// GAU_20873541058763
// MI455X (gfx1250) — matrix operand order repaired and checked by reading, not compiled, not run on hardware
//
#include <hip/hip_runtime.h>
#include <hip/hip_bf16.h>

// ---------------------------------------------------------------------------
// GAU layer for MI455X (gfx1250): bf16 WMMA (f32 accum) for all four GEMMs.
// All GEMMs consume B as [N][K] row-major so both LDS stages are b128 copies.
// Double-buffered LDS: one barrier per K-step, next tile's global loads
// overlap the current tile's WMMA stream.
// ---------------------------------------------------------------------------

typedef __attribute__((ext_vector_type(16))) __bf16 v16bf;
typedef __attribute__((ext_vector_type(8)))  float  v8f;

#define HID    768
#define E_DIM  1536
#define S_DIM  128
#define LSEQ   512
#define BATCH  32
#define NTOK   (BATCH * LSEQ)      // 16384
#define NUV    (2 * E_DIM + S_DIM) // 3200

#define BM  256
#define BN  128
#define BK  32
#define STA 40   // LDS row stride (elements) = BK + 8 pad, keeps 16B alignment
#define STB 40
#define SZA (BM * STA)   // one A buffer, elements
#define SZB (BN * STB)   // one B buffer, elements

union FragU {
    uint4  u4[2];
    v16bf  v;
    __bf16 h[16];
};
union Pack8 {
    uint4  u;
    __bf16 h[8];
};

// ---------------------------------------------------------------------------
// Tiled GEMM core: C(BM x BN) += A(BM x K) * B^T, with B stored [N][K]
// row-major. 256 threads = 8 waves, wave grid 4(m) x 2(n), wave tile 64x64 =
// 4x4 WMMA accumulators. LDS is double buffered (sA/sB hold 2 tiles each).
// ---------------------------------------------------------------------------
__device__ __forceinline__ void gemm_core(const __bf16* __restrict__ A, int lda,
                                          const __bf16* __restrict__ B, int ldb,
                                          int K, __bf16* sA, __bf16* sB,
                                          v8f acc[4][4])
{
    const int t    = threadIdx.x;
    const int lane = t & 31;
    const int wave = t >> 5;
    const int wm   = (wave & 3) * 64;
    const int wn   = (wave >> 2) * 64;
    const int brow = t >> 1;        // 0..127 : B tile N-row
    const int bks  = (t & 1) * 16;  // 0 | 16 : B K sub-chunk

    const int m0 = wm + (lane & 15);
    const int kA = (lane >> 4) * 8;   // lanes 0-15: K 0..7/16..23; 16-31: 8..15/24..31
    const int n0 = wn + (lane & 15);
    const int kB = (lane >> 4) * 8;   // lanes 0-15: K 0..7/16..23; 16-31: 8..15/24..31

    // ---- preload tile 0 into registers ------------------------------------
    uint4 a0, a1, a2, a3, b0, b1;
    {
        const uint4* ag = (const uint4*)(A + (size_t)t * lda);
        a0 = ag[0]; a1 = ag[1]; a2 = ag[2]; a3 = ag[3];
        const uint4* bg = (const uint4*)(B + (size_t)brow * ldb + bks);
        b0 = bg[0]; b1 = bg[1];
    }

    int buf = 0;
    for (int k0 = 0; k0 < K; k0 += BK) {
        __bf16* cA = sA + buf * SZA;
        __bf16* cB = sB + buf * SZB;
        // ---- commit staged registers to current LDS buffer ----------------
        {
            uint4* as = (uint4*)(cA + t * STA);
            as[0] = a0; as[1] = a1; as[2] = a2; as[3] = a3;
            uint4* bs = (uint4*)(cB + brow * STB + bks);
            bs[0] = b0; bs[1] = b1;
        }
        __syncthreads();   // single barrier per K-step (double buffering)

        // ---- issue next tile's global loads (overlap with WMMAs) ----------
        if (k0 + BK < K) {
            const uint4* ag = (const uint4*)(A + (size_t)t * lda + k0 + BK);
            a0 = ag[0]; a1 = ag[1]; a2 = ag[2]; a3 = ag[3];
            const uint4* bg = (const uint4*)(B + (size_t)brow * ldb + k0 + BK + bks);
            b0 = bg[0]; b1 = bg[1];
        }

        // ---- fragments per ISA 16-bit A 16x32 / B 32x16 layouts -----------
        FragU afr[4], bfr[4];
#pragma unroll
        for (int i = 0; i < 4; ++i) {
            const __bf16* p = cA + (size_t)(m0 + i * 16) * STA + kA;
            afr[i].u4[0] = *(const uint4*)(p);
            afr[i].u4[1] = *(const uint4*)(p + 16);
        }
#pragma unroll
        for (int j = 0; j < 4; ++j) {
            const __bf16* p = cB + (size_t)(n0 + j * 16) * STB + kB;
            bfr[j].u4[0] = *(const uint4*)(p);
            bfr[j].u4[1] = *(const uint4*)(p + 16);
        }
#pragma unroll
        for (int i = 0; i < 4; ++i)
#pragma unroll
            for (int j = 0; j < 4; ++j)
                acc[i][j] = __builtin_amdgcn_wmma_f32_16x16x32_bf16(
                    false, afr[i].v, false, bfr[j].v,
                    (short)0, acc[i][j], false, false);
        buf ^= 1;
    }
}

// C/D f32 16x16 layout: VGPR r, lane l -> row = r + 8*(l>>4), col = l&15
#define EPILOGUE_LOOP(ROW, COL, VAL, BODY)                                     \
    {                                                                          \
        const int lane = threadIdx.x & 31;                                     \
        const int wave = threadIdx.x >> 5;                                     \
        const int wm   = (wave & 3) * 64;                                      \
        const int wn   = (wave >> 2) * 64;                                     \
        _Pragma("unroll") for (int i = 0; i < 4; ++i)                          \
        _Pragma("unroll") for (int j = 0; j < 4; ++j)                          \
        _Pragma("unroll") for (int r = 0; r < 8; ++r) {                        \
            const int ROW = wm + i * 16 + r + ((lane >> 4) << 3);              \
            const int COL = wn + j * 16 + (lane & 15);                         \
            float VAL     = acc[i][j][r];                                      \
            BODY                                                               \
        }                                                                      \
    }

// ---------------------------------------------------------------------------
// Kernels
// ---------------------------------------------------------------------------

// f32 [R][C] -> bf16 transposed [C][R], LDS-tiled 32x32, both sides coalesced.
__global__ __launch_bounds__(256) void k_cvt_t(const float* __restrict__ src,
                                               __bf16* __restrict__ dst,
                                               int R, int C)
{
    __shared__ __bf16 tile[32][33];
    const int c0 = blockIdx.x * 32;
    const int r0 = blockIdx.y * 32;
    const int tx = threadIdx.x & 31;
    const int ty = threadIdx.x >> 5;   // 0..7
#pragma unroll
    for (int rr = ty; rr < 32; rr += 8)
        tile[rr][tx] = (__bf16)src[(size_t)(r0 + rr) * C + (c0 + tx)];
    __syncthreads();
#pragma unroll
    for (int cc = ty; cc < 32; cc += 8)
        dst[(size_t)(c0 + cc) * R + (r0 + tx)] = tile[tx][cc];
}

__global__ __launch_bounds__(256) void k_layernorm(const float* __restrict__ x,
                                                   const float* __restrict__ g,
                                                   const float* __restrict__ be,
                                                   __bf16* __restrict__ xn)
{
    __shared__ float s1[256], s2[256];
    const int row   = blockIdx.x;
    const int t     = threadIdx.x;
    const float* xp = x + (size_t)row * HID;
    float a0 = xp[t], a1 = xp[t + 256], a2 = xp[t + 512];
    s1[t] = a0 + a1 + a2;
    s2[t] = a0 * a0 + a1 * a1 + a2 * a2;
    __syncthreads();
    for (int off = 128; off > 0; off >>= 1) {
        if (t < off) { s1[t] += s1[t + off]; s2[t] += s2[t + off]; }
        __syncthreads();
    }
    const float mu  = s1[0] * (1.0f / HID);
    const float var = s2[0] * (1.0f / HID) - mu * mu;
    const float rs  = rsqrtf(var + 1e-5f);
    __bf16* op = xn + (size_t)row * HID;
    op[t]       = (__bf16)((a0 - mu) * rs * g[t]       + be[t]);
    op[t + 256] = (__bf16)((a1 - mu) * rs * g[t + 256] + be[t + 256]);
    op[t + 512] = (__bf16)((a2 - mu) * rs * g[t + 512] + be[t + 512]);
}

// uv = silu(xn @ W_uv + uv_b); u,v written TRANSPOSED [b][E][L] (packed b128
// stores: the 8 accumulator rows per lane are consecutive in M); base -> f32.
__global__ __launch_bounds__(256) void k_gemm_uv(const __bf16* __restrict__ xn,
                                                 const __bf16* __restrict__ WuvT,
                                                 const float* __restrict__ uvb,
                                                 __bf16* __restrict__ ut,
                                                 __bf16* __restrict__ vt,
                                                 float* __restrict__ baseo)
{
    __shared__ __align__(16) __bf16 sA[2 * SZA];
    __shared__ __align__(16) __bf16 sB[2 * SZB];
    const int blkM = blockIdx.x * BM;
    const int blkN = blockIdx.y * BN;
    v8f acc[4][4] = {};
    gemm_core(xn + (size_t)blkM * HID, HID, WuvT + (size_t)blkN * HID, HID,
              HID, sA, sB, acc);

    const int lane = threadIdx.x & 31;
    const int wave = threadIdx.x >> 5;
    const int wm   = (wave & 3) * 64;
    const int wn   = (wave >> 2) * 64;
#pragma unroll
    for (int i = 0; i < 4; ++i) {
        const int mbase = blkM + wm + i * 16 + ((lane >> 4) << 3); // mult of 8
        const int b     = mbase >> 9;
        const int ml    = mbase & (LSEQ - 1);
#pragma unroll
        for (int j = 0; j < 4; ++j) {
            const int   col  = blkN + wn + j * 16 + (lane & 15);
            const float bias = uvb[col];
            if (col < 2 * E_DIM) {
                Pack8 pk;
#pragma unroll
                for (int r = 0; r < 8; ++r) {
                    float s = acc[i][j][r] + bias;
                    s = s / (1.0f + __expf(-s));   // silu
                    pk.h[r] = (__bf16)s;
                }
                __bf16* dst = (col < E_DIM)
                    ? (ut + ((size_t)b * E_DIM + col) * LSEQ + ml)
                    : (vt + ((size_t)b * E_DIM + (col - E_DIM)) * LSEQ + ml);
                *(uint4*)dst = pk.u;
            } else {
                const int sc = col - 2 * E_DIM;
#pragma unroll
                for (int r = 0; r < 8; ++r) {
                    float s = acc[i][j][r] + bias;
                    s = s / (1.0f + __expf(-s));
                    baseo[(size_t)(mbase + r) * S_DIM + sc] = s;
                }
            }
        }
    }
}

// base -> (gamma,beta scale) -> RoPE -> q,k (bf16, [token][S])
__global__ void k_rope(const float* __restrict__ base,
                       const float* __restrict__ gq,
                       const float* __restrict__ bq,
                       __bf16* __restrict__ q, __bf16* __restrict__ k)
{
    const int tok = blockIdx.x;          // 0..16383
    const int n   = tok & (LSEQ - 1);    // seq position
    const int d   = threadIdx.x;         // 0..63
    const float* bp = base + (size_t)tok * S_DIM;
    const float inv = __powf(10000.0f, -(float)d * (1.0f / 64.0f));
    const float ang = (float)n * inv;
    const float sn = __sinf(ang), cs = __cosf(ang);
    {   // q row (gamma_qk[0], beta_qk[0])
        float x1 = bp[d]      * gq[d]      + bq[d];
        float x2 = bp[d + 64] * gq[d + 64] + bq[d + 64];
        q[(size_t)tok * S_DIM + d]      = (__bf16)(x1 * cs - x2 * sn);
        q[(size_t)tok * S_DIM + d + 64] = (__bf16)(x2 * cs + x1 * sn);
    }
    {   // k row (gamma_qk[1], beta_qk[1])
        const float* g1 = gq + S_DIM;
        const float* b1 = bq + S_DIM;
        float x1 = bp[d]      * g1[d]      + b1[d];
        float x2 = bp[d + 64] * g1[d + 64] + b1[d + 64];
        k[(size_t)tok * S_DIM + d]      = (__bf16)(x1 * cs - x2 * sn);
        k[(size_t)tok * S_DIM + d + 64] = (__bf16)(x2 * cs + x1 * sn);
    }
}

// kernel = relu(q k^T / L + bias)^2  (per batch), bf16 out [n][m]
__global__ __launch_bounds__(256) void k_qk(const __bf16* __restrict__ q,
                                            const __bf16* __restrict__ kmat,
                                            const float* __restrict__ wrel,
                                            __bf16* __restrict__ ker)
{
    __shared__ __align__(16) __bf16 sA[2 * SZA];
    __shared__ __align__(16) __bf16 sB[2 * SZB];
    const int b    = blockIdx.z;
    const int blkM = blockIdx.x * BM;
    const int blkN = blockIdx.y * BN;
    const __bf16* A = q    + (size_t)b * LSEQ * S_DIM + (size_t)blkM * S_DIM;
    const __bf16* B = kmat + (size_t)b * LSEQ * S_DIM + (size_t)blkN * S_DIM;
    v8f acc[4][4] = {};
    gemm_core(A, S_DIM, B, S_DIM, S_DIM, sA, sB, acc);
    __bf16* kb = ker + (size_t)b * LSEQ * LSEQ;
    EPILOGUE_LOOP(rr, cc, val, {
        const int row = blkM + rr;   // n
        const int col = blkN + cc;   // m
        float s = val * (1.0f / (float)LSEQ) + wrel[col - row + (LSEQ - 1)];
        s = fmaxf(s, 0.0f);
        kb[(size_t)row * LSEQ + col] = (__bf16)(s * s);
    })
}

// att = u * (kernel @ v)  (per batch); vt/ut are [b][E][L]; att row-major.
__global__ __launch_bounds__(256) void k_pv(const __bf16* __restrict__ ker,
                                            const __bf16* __restrict__ vt,
                                            const __bf16* __restrict__ ut,
                                            __bf16* __restrict__ att)
{
    __shared__ __align__(16) __bf16 sA[2 * SZA];
    __shared__ __align__(16) __bf16 sB[2 * SZB];
    const int b    = blockIdx.z;
    const int blkM = blockIdx.x * BM;
    const int blkN = blockIdx.y * BN;
    const __bf16* A = ker + (size_t)b * LSEQ * LSEQ + (size_t)blkM * LSEQ;
    const __bf16* B = vt  + (size_t)b * E_DIM * LSEQ + (size_t)blkN * LSEQ;
    v8f acc[4][4] = {};
    gemm_core(A, LSEQ, B, LSEQ, LSEQ, sA, sB, acc);

    const int lane = threadIdx.x & 31;
    const int wave = threadIdx.x >> 5;
    const int wm   = (wave & 3) * 64;
    const int wn   = (wave >> 2) * 64;
#pragma unroll
    for (int i = 0; i < 4; ++i) {
        const int ml = blkM + wm + i * 16 + ((lane >> 4) << 3); // mult of 8
#pragma unroll
        for (int j = 0; j < 4; ++j) {
            const int col = blkN + wn + j * 16 + (lane & 15);
            Pack8 uu;
            uu.u = *(const uint4*)(ut + ((size_t)b * E_DIM + col) * LSEQ + ml);
#pragma unroll
            for (int r = 0; r < 8; ++r) {
                const float s = acc[i][j][r] * (float)uu.h[r];
                att[((size_t)b * LSEQ + ml + r) * E_DIM + col] = (__bf16)s;
            }
        }
    }
}

// out = att @ W_o + o_b + shortcut (f32 out)
__global__ __launch_bounds__(256) void k_out(const __bf16* __restrict__ att,
                                             const __bf16* __restrict__ WoT,
                                             const float* __restrict__ ob,
                                             const float* __restrict__ x,
                                             float* __restrict__ out)
{
    __shared__ __align__(16) __bf16 sA[2 * SZA];
    __shared__ __align__(16) __bf16 sB[2 * SZB];
    const int blkM = blockIdx.x * BM;
    const int blkN = blockIdx.y * BN;
    v8f acc[4][4] = {};
    gemm_core(att + (size_t)blkM * E_DIM, E_DIM,
              WoT + (size_t)blkN * E_DIM, E_DIM, E_DIM, sA, sB, acc);
    EPILOGUE_LOOP(rr, cc, val, {
        const size_t row = (size_t)blkM + rr;
        const int    col = blkN + cc;
        out[row * HID + col] = val + ob[col] + x[row * HID + col];
    })
}

// ---------------------------------------------------------------------------
// Launcher
// ---------------------------------------------------------------------------
extern "C" void kernel_launch(void* const* d_in, const int* in_sizes, int n_in,
                              void* d_out, int out_size, void* d_ws, size_t ws_size,
                              hipStream_t stream)
{
    (void)in_sizes; (void)n_in; (void)out_size; (void)ws_size;
    const float* x     = (const float*)d_in[0];
    const float* ln_g  = (const float*)d_in[1];
    const float* ln_b  = (const float*)d_in[2];
    const float* uv_W  = (const float*)d_in[3];
    const float* uv_b  = (const float*)d_in[4];
    const float* g_qk  = (const float*)d_in[5];
    const float* b_qk  = (const float*)d_in[6];
    const float* w_rel = (const float*)d_in[7];
    const float* o_W   = (const float*)d_in[8];
    const float* o_b   = (const float*)d_in[9];
    float*       out   = (float*)d_out;

    char*  wsb = (char*)d_ws;
    size_t off = 0;
    auto nxt = [&](size_t bytes) {
        void* p = wsb + off;
        off += (bytes + 255) & ~(size_t)255;
        return p;
    };
    __bf16* xn   = (__bf16*)nxt((size_t)NTOK * HID * 2);
    __bf16* wuvT = (__bf16*)nxt((size_t)HID * NUV * 2);    // [NUV][HID]
    __bf16* woT  = (__bf16*)nxt((size_t)E_DIM * HID * 2);  // [HID][E]
    __bf16* ut   = (__bf16*)nxt((size_t)NTOK * E_DIM * 2); // [b][E][L]
    __bf16* vt   = (__bf16*)nxt((size_t)NTOK * E_DIM * 2); // [b][E][L]
    float*  basf = (float*) nxt((size_t)NTOK * S_DIM * 4);
    __bf16* qb   = (__bf16*)nxt((size_t)NTOK * S_DIM * 2);
    __bf16* kb   = (__bf16*)nxt((size_t)NTOK * S_DIM * 2);
    __bf16* ker  = (__bf16*)nxt((size_t)BATCH * LSEQ * LSEQ * 2);
    __bf16* att  = (__bf16*)nxt((size_t)NTOK * E_DIM * 2);

    // Weight conversion + transpose: W_uv [HID][NUV] -> [NUV][HID],
    //                                W_o  [E][HID]   -> [HID][E]
    k_cvt_t<<<dim3(NUV / 32, HID / 32), 256, 0, stream>>>(uv_W, wuvT, HID, NUV);
    k_cvt_t<<<dim3(HID / 32, E_DIM / 32), 256, 0, stream>>>(o_W, woT, E_DIM, HID);

    k_layernorm<<<NTOK, 256, 0, stream>>>(x, ln_g, ln_b, xn);

    k_gemm_uv<<<dim3(NTOK / BM, NUV / BN), 256, 0, stream>>>(xn, wuvT, uv_b,
                                                             ut, vt, basf);

    k_rope<<<NTOK, 64, 0, stream>>>(basf, g_qk, b_qk, qb, kb);

    k_qk<<<dim3(LSEQ / BM, LSEQ / BN, BATCH), 256, 0, stream>>>(qb, kb, w_rel, ker);

    k_pv<<<dim3(LSEQ / BM, E_DIM / BN, BATCH), 256, 0, stream>>>(ker, vt, ut, att);

    k_out<<<dim3(NTOK / BM, HID / BN), 256, 0, stream>>>(att, woT, o_b, x, out);
}
